// Net_66821101191726
// MI455X (gfx1250) — compile-verified, matrix operand order checked by reading, not run on hardware
//
#include <hip/hip_runtime.h>
#include <hip/hip_bf16.h>
#include <math.h>

// ---------- vector types for WMMA ----------
typedef __attribute__((ext_vector_type(16))) __bf16 v16bf;
typedef __attribute__((ext_vector_type(8)))  __bf16 v8bf;
typedef __attribute__((ext_vector_type(8)))  float  v8f;

// =====================================================================
// Zero the 5 entropy accumulator slots at the tail of d_out
// =====================================================================
__global__ void zero_h_kernel(float* h) {
    if (threadIdx.x < 5) h[threadIdx.x] = 0.0f;
}

// =====================================================================
// f32 -> bf16 weight conversion (weights stay L2-resident afterwards)
// =====================================================================
__global__ void cvt_f32_bf16_kernel(const float* __restrict__ src,
                                    __bf16* __restrict__ dst, int n) {
    int i = blockIdx.x * 256 + threadIdx.x;
    if (i < n) dst[i] = (__bf16)src[i];
}

// =====================================================================
// Fused GEMM + bias + leaky-ReLU using v_wmma_f32_16x16x32_bf16
//   Y[M,N] = leaky_relu(X[M,K] @ Wb[N,K]^T + bias[N])
//
// - 8-element (16B) chunked staging: global_load_b128 -> regs ->
//   cvt_pk_bf16 -> single ds_store_b128. K is a multiple of 8 in every
//   layer, so chunk-level predication is exact (no per-element guards).
// - Register + LDS ping-pong double buffering: loads for k-block t+1
//   are issued before the WMMAs of block t; the loadcnt wait lands
//   after the matrix math.
// - Fragment layout per CDNA5 16-bit operand spec: lanes 0-15 hold
//   row=lane with K 0-7 / 16-23, lanes 16-31 hold K 8-15 / 24-31.
// =====================================================================
template <int BM, int BN, int WAVES_M, int WAVES_N>
__global__ __launch_bounds__(WAVES_M * WAVES_N * 32)
void gemm_bf16_leaky(const float* __restrict__ X,
                     const __bf16* __restrict__ Wb,
                     const float* __restrict__ bias,
                     float* __restrict__ Y,
                     int M, int N, int K, float alpha)
{
    constexpr int NT  = WAVES_M * WAVES_N * 32;
    constexpr int WM  = BM / WAVES_M;
    constexpr int WN  = BN / WAVES_N;
    constexpr int FM  = WM / 16;
    constexpr int FN  = WN / 16;
    constexpr int ACH = (BM * 4 + NT - 1) / NT;   // 8-elem chunks of A per thread
    constexpr int BCH = (BN * 4 + NT - 1) / NT;   // 8-elem chunks of B per thread
    constexpr bool A_FULL = (BM * 4 % NT) == 0;
    constexpr bool B_FULL = (BN * 4 % NT) == 0;

    __shared__ __bf16 lds_a[2][BM * 32];
    __shared__ __bf16 lds_b[2][BN * 32];

    const int tid     = threadIdx.x;
    const int wave    = tid >> 5;
    const int lane    = tid & 31;
    const int lane_lo = lane & 15;
    const int half    = lane >> 4;
    const int wm      = wave % WAVES_M;
    const int wn      = wave / WAVES_M;
    const int blockM  = blockIdx.x * BM;
    const int blockN  = blockIdx.y * BN;

    v8f acc[FM][FN];
#pragma unroll
    for (int i = 0; i < FM; ++i)
#pragma unroll
        for (int j = 0; j < FN; ++j)
#pragma unroll
            for (int r = 0; r < 8; ++r) acc[i][j][r] = 0.0f;

    float4 areg[ACH][2];
    uint4  breg[BCH];

    // ---- staging helpers (chunk = 8 consecutive K at one row) ----
    auto load_tile = [&](int k0) {
#pragma unroll
        for (int i = 0; i < ACH; ++i) {
            const int c  = tid + i * NT;
            const int rr = c >> 2, kk = (c & 3) * 8;
            const int kg = k0 + kk;
            float4 z = {0.0f, 0.0f, 0.0f, 0.0f};
            areg[i][0] = z; areg[i][1] = z;
            if ((A_FULL || c < BM * 4) && kg < K) {
                const float4* src =
                    (const float4*)(X + (size_t)(blockM + rr) * K + kg);
                areg[i][0] = src[0];
                areg[i][1] = src[1];
            }
        }
#pragma unroll
        for (int i = 0; i < BCH; ++i) {
            const int c  = tid + i * NT;
            const int rr = c >> 2, kk = (c & 3) * 8;
            const int kg = k0 + kk;
            uint4 z = {0u, 0u, 0u, 0u};
            breg[i] = z;
            if ((B_FULL || c < BN * 4) && kg < K)
                breg[i] = *(const uint4*)(Wb + (size_t)(blockN + rr) * K + kg);
        }
    };
    auto store_tile = [&](int buf) {
#pragma unroll
        for (int i = 0; i < ACH; ++i) {
            const int c = tid + i * NT;
            if (A_FULL || c < BM * 4) {
                const int rr = c >> 2, kk = (c & 3) * 8;
                v8bf pk;
                pk[0] = (__bf16)areg[i][0].x; pk[1] = (__bf16)areg[i][0].y;
                pk[2] = (__bf16)areg[i][0].z; pk[3] = (__bf16)areg[i][0].w;
                pk[4] = (__bf16)areg[i][1].x; pk[5] = (__bf16)areg[i][1].y;
                pk[6] = (__bf16)areg[i][1].z; pk[7] = (__bf16)areg[i][1].w;
                *(v8bf*)&lds_a[buf][rr * 32 + kk] = pk;
            }
        }
#pragma unroll
        for (int i = 0; i < BCH; ++i) {
            const int c = tid + i * NT;
            if (B_FULL || c < BN * 4) {
                const int rr = c >> 2, kk = (c & 3) * 8;
                *(uint4*)&lds_b[buf][rr * 32 + kk] = breg[i];
            }
        }
    };

    const int T = (K + 31) / 32;

    // prologue: stage tile 0
    load_tile(0);
    store_tile(0);
    __syncthreads();

    for (int t = 0; t < T; ++t) {
        const int  buf  = t & 1;
        const bool more = (t + 1 < T);

        // issue next tile's global loads before the math
        if (more) load_tile((t + 1) * 32);

        // ---- build fragments from LDS ----
        v16bf afrag[FM], bfrag[FN];
#pragma unroll
        for (int fm = 0; fm < FM; ++fm) {
            const __bf16* p = &lds_a[buf][(wm * WM + fm * 16 + lane_lo) * 32];
            v8bf lo = *(const v8bf*)(p + half * 8);
            v8bf hi = *(const v8bf*)(p + 16 + half * 8);
#pragma unroll
            for (int e = 0; e < 8; ++e) { afrag[fm][e] = lo[e]; afrag[fm][8 + e] = hi[e]; }
        }
#pragma unroll
        for (int fn = 0; fn < FN; ++fn) {
            const __bf16* p = &lds_b[buf][(wn * WN + fn * 16 + lane_lo) * 32];
            v8bf lo = *(const v8bf*)(p + half * 8);
            v8bf hi = *(const v8bf*)(p + 16 + half * 8);
#pragma unroll
            for (int e = 0; e < 8; ++e) { bfrag[fn][e] = lo[e]; bfrag[fn][8 + e] = hi[e]; }
        }

        // ---- WMMA accumulate ----
#pragma unroll
        for (int fm = 0; fm < FM; ++fm)
#pragma unroll
            for (int fn = 0; fn < FN; ++fn)
                acc[fm][fn] = __builtin_amdgcn_wmma_f32_16x16x32_bf16(
                    false, afrag[fm], false, bfrag[fn],
                    (short)0, acc[fm][fn], false, false);

        // store next tile into the other LDS buffer (waits for loads here,
        // i.e. after the WMMAs have been issued)
        if (more) store_tile(buf ^ 1);
        __syncthreads();
    }

    // ---- epilogue: bias + leaky-ReLU (grid divides exactly; no guards) ----
#pragma unroll
    for (int fm = 0; fm < FM; ++fm) {
#pragma unroll
        for (int fn = 0; fn < FN; ++fn) {
            const int col = blockN + wn * WN + fn * 16 + lane_lo;
            const float b = bias[col];
#pragma unroll
            for (int r = 0; r < 8; ++r) {
                const int row = blockM + wm * WM + fm * 16 + half * 8 + r;
                float v = acc[fm][fn][r] + b;
                v = (v > 0.0f) ? v : alpha * v;
                Y[(size_t)row * N + col] = v;
            }
        }
    }
}

// =====================================================================
// Per-row histogram entropy. One wave32 per row, 8 rows per block.
// H_row = -sum p log p + log(D), accumulated as mean via global f32 atomic.
// =====================================================================
template <int D>
__global__ __launch_bounds__(256)
void entropy_kernel(const float* __restrict__ A, float* __restrict__ Hout,
                    float invB)
{
    __shared__ unsigned cnt[8][D];
    const int wave = threadIdx.x >> 5;
    const int lane = threadIdx.x & 31;
    const int row  = blockIdx.x * 8 + wave;

    constexpr int PER = D / 32;
    float v[PER];
    const float* p = A + (size_t)row * D;
#pragma unroll
    for (int j = 0; j < PER; ++j) v[j] = p[lane + j * 32];

    float mn = v[0], mx = v[0];
#pragma unroll
    for (int j = 1; j < PER; ++j) { mn = fminf(mn, v[j]); mx = fmaxf(mx, v[j]); }
#pragma unroll
    for (int off = 16; off > 0; off >>= 1) {
        mn = fminf(mn, __shfl_xor(mn, off, 32));
        mx = fmaxf(mx, __shfl_xor(mx, off, 32));
    }
    float w     = mx - mn;
    float scale = (float)D / ((w > 0.0f) ? w : 1.0f);

    unsigned* c = cnt[wave];
    for (int i = lane; i < D; i += 32) c[i] = 0u;
    // DS ops from one wave complete in order (DScnt) -> init visible.
#pragma unroll
    for (int j = 0; j < PER; ++j) {
        int idx = (int)floorf((v[j] - mn) * scale);
        idx = (idx < 0) ? 0 : ((idx > D - 1) ? D - 1 : idx);
        atomicAdd(&c[idx], 1u);
    }

    float s = 0.0f;
    for (int i = lane; i < D; i += 32) {
        unsigned cc = c[i];
        if (cc) {
            float pp = (float)cc * (1.0f / (float)D);
            s -= pp * logf(pp);
        }
    }
#pragma unroll
    for (int off = 16; off > 0; off >>= 1) s += __shfl_xor(s, off, 32);
    if (lane == 0) atomicAdd(Hout, (s + logf((float)D)) * invB);
}

// =====================================================================
// Head: z = relu(a @ W6^T + b6) ; out = log_softmax(z). One thread/row.
// =====================================================================
__global__ __launch_bounds__(256)
void head_kernel(const float* __restrict__ A,   // [B,32]
                 const float* __restrict__ W6,  // [10,32]
                 const float* __restrict__ b6,  // [10]
                 float* __restrict__ out)       // [B,10]
{
    __shared__ float w[320];
    __shared__ float bb[10];
    for (int i = threadIdx.x; i < 320; i += 256) w[i] = W6[i];
    if (threadIdx.x < 10) bb[threadIdx.x] = b6[threadIdx.x];
    __syncthreads();

    const int row = blockIdx.x * 256 + threadIdx.x;
    float a[32];
    const float4* ap = (const float4*)(A + (size_t)row * 32);
#pragma unroll
    for (int j = 0; j < 8; ++j) {
        float4 t = ap[j];
        a[4 * j + 0] = t.x; a[4 * j + 1] = t.y; a[4 * j + 2] = t.z; a[4 * j + 3] = t.w;
    }
    float z[10];
#pragma unroll
    for (int n = 0; n < 10; ++n) {
        float s = bb[n];
#pragma unroll
        for (int k = 0; k < 32; ++k) s = fmaf(a[k], w[n * 32 + k], s);
        z[n] = fmaxf(s, 0.0f);
    }
    float m = z[0];
#pragma unroll
    for (int n = 1; n < 10; ++n) m = fmaxf(m, z[n]);
    float sum = 0.0f;
#pragma unroll
    for (int n = 0; n < 10; ++n) sum += expf(z[n] - m);
    float lse = m + logf(sum);
    float* o = out + (size_t)row * 10;
#pragma unroll
    for (int n = 0; n < 10; ++n) o[n] = z[n] - lse;
}

// =====================================================================
// Launch
// =====================================================================
extern "C" void kernel_launch(void* const* d_in, const int* in_sizes, int n_in,
                              void* d_out, int out_size, void* d_ws, size_t ws_size,
                              hipStream_t stream)
{
    const int B = 65536;
    const float ALPHA = 0.01f;
    const float invB  = 1.0f / (float)B;

    const float* x  = (const float*)d_in[0];
    const float* W1 = (const float*)d_in[1];  const float* b1 = (const float*)d_in[2];
    const float* W2 = (const float*)d_in[3];  const float* b2 = (const float*)d_in[4];
    const float* W3 = (const float*)d_in[5];  const float* b3 = (const float*)d_in[6];
    const float* W4 = (const float*)d_in[7];  const float* b4 = (const float*)d_in[8];
    const float* W5 = (const float*)d_in[9];  const float* b5 = (const float*)d_in[10];
    const float* W6 = (const float*)d_in[11]; const float* b6 = (const float*)d_in[12];

    float* out = (float*)d_out;
    float* H   = out + (out_size - 5);   // H1..H5 scalars at tail

    // ---- workspace layout ----
    char* ws = (char*)d_ws;
    __bf16* wb1 = (__bf16*)(ws + 0);                 // 256*784
    __bf16* wb2 = (__bf16*)(ws + 401408);            // 128*256
    __bf16* wb3 = (__bf16*)(ws + 466944);            // 128*128
    __bf16* wb4 = (__bf16*)(ws + 499712);            // 128*128
    __bf16* wb5 = (__bf16*)(ws + 532480);            // 32*128
    float*  buf0 = (float*)(ws + (1u << 20));        // 65536*256 f32 (67.1 MB)
    float*  buf1 = (float*)(ws + (1u << 20) + (size_t)B * 256 * 4); // 65536*128 f32

    // 0) zero entropy accumulators
    zero_h_kernel<<<1, 32, 0, stream>>>(H);

    // 1) weights -> bf16
    cvt_f32_bf16_kernel<<<(256 * 784 + 255) / 256, 256, 0, stream>>>(W1, wb1, 256 * 784);
    cvt_f32_bf16_kernel<<<(128 * 256 + 255) / 256, 256, 0, stream>>>(W2, wb2, 128 * 256);
    cvt_f32_bf16_kernel<<<(128 * 128 + 255) / 256, 256, 0, stream>>>(W3, wb3, 128 * 128);
    cvt_f32_bf16_kernel<<<(128 * 128 + 255) / 256, 256, 0, stream>>>(W4, wb4, 128 * 128);
    cvt_f32_bf16_kernel<<<(32 * 128 + 255) / 256, 256, 0, stream>>>(W5, wb5, 32 * 128);

    // 2) layer 1: [B,784] x [256,784]^T -> buf0 [B,256]
    gemm_bf16_leaky<128, 64, 4, 2><<<dim3(B / 128, 256 / 64), 256, 0, stream>>>(
        x, wb1, b1, buf0, B, 256, 784, ALPHA);
    entropy_kernel<256><<<B / 8, 256, 0, stream>>>(buf0, H + 0, invB);

    // 3) layer 2: buf0 [B,256] -> buf1 [B,128]
    gemm_bf16_leaky<128, 64, 4, 2><<<dim3(B / 128, 128 / 64), 256, 0, stream>>>(
        buf0, wb2, b2, buf1, B, 128, 256, ALPHA);
    entropy_kernel<128><<<B / 8, 256, 0, stream>>>(buf1, H + 1, invB);

    // 4) layer 3: buf1 -> buf0 [B,128]
    gemm_bf16_leaky<128, 64, 4, 2><<<dim3(B / 128, 128 / 64), 256, 0, stream>>>(
        buf1, wb3, b3, buf0, B, 128, 128, ALPHA);
    entropy_kernel<128><<<B / 8, 256, 0, stream>>>(buf0, H + 2, invB);

    // 5) layer 4: buf0 -> buf1 [B,128]
    gemm_bf16_leaky<128, 64, 4, 2><<<dim3(B / 128, 128 / 64), 256, 0, stream>>>(
        buf0, wb4, b4, buf1, B, 128, 128, ALPHA);
    entropy_kernel<128><<<B / 8, 256, 0, stream>>>(buf1, H + 3, invB);

    // 6) layer 5: buf1 -> buf0 [B,32]
    gemm_bf16_leaky<128, 32, 8, 1><<<dim3(B / 128, 1), 256, 0, stream>>>(
        buf1, wb5, b5, buf0, B, 32, 128, ALPHA);
    entropy_kernel<32><<<B / 8, 256, 0, stream>>>(buf0, H + 4, invB);

    // 7) head: buf0 [B,32] -> out [B,10] log-softmax
    head_kernel<<<B / 256, 256, 0, stream>>>(buf0, W6, b6, out);
}
